// AttentionRoPE_70007966925376
// MI455X (gfx1250) — matrix operand order repaired and checked by reading, not compiled, not run on hardware
//
#include <hip/hip_runtime.h>
#include <hip/hip_bf16.h>

// MI455X / gfx1250 fused attention+RoPE, bf16 WMMA pipeline:
//   cvt(x,w) -> [QKV GEMM + RMSnorm + RoPE] -> flash attention (S^T trick) -> proj GEMM + bias
// Shared tiles (weights, K, V) staged via GLOBAL_LOAD_ASYNC_TO_LDS_B128,
// double-buffered, synced with s_wait_asynccnt + workgroup barrier.
// Wave tiles sized for ~1 fragment load per WMMA.

typedef __bf16 bf16_t;
typedef __attribute__((ext_vector_type(16))) __bf16       v16bf;
typedef __attribute__((ext_vector_type(8)))  float        v8f;
typedef __attribute__((ext_vector_type(4)))  unsigned int u32x4;
typedef __attribute__((ext_vector_type(2)))  unsigned int u32x2;
typedef __attribute__((ext_vector_type(4)))  float        f32x4;

namespace {
constexpr int kC  = 1024;
constexpr int kH  = 16;
constexpr int kHd = 64;
constexpr int kN  = 2048;
constexpr int kM  = 4096;       // B*N flattened rows
constexpr int k3C = 3 * kC;
}

__device__ __forceinline__ v8f wmma_bf16(v16bf a, v16bf b, v8f c) {
  return __builtin_amdgcn_wmma_f32_16x16x32_bf16(false, a, false, b, (short)0, c,
                                                 false, false);
}

union FragU { u32x4 u[2]; v16bf v; };

// A fragment (16x32 bf16, M rows of row-major src). lanes<16: K=k0+0..7,16..23;
// lanes>=16: K=k0+8..15,24..31. M = row0 + (lane&15).
__device__ __forceinline__ v16bf load_a_frag(const bf16_t* base, int row0, int ldm,
                                             int k0, int lane) {
  int r  = row0 + (lane & 15);
  int ko = k0 + ((lane & 16) ? 8 : 0);
  const bf16_t* p = base + (size_t)r * ldm + ko;
  FragU f;
  f.u[0] = *(const u32x4*)(p);
  f.u[1] = *(const u32x4*)(p + 16);
  return f.v;
}

// B fragment (32x16 bf16) where B[k][n] = SRC[n][k], SRC row-major (ldm).
// lanes<16: K=k0+0..7,16..23 at N=lane; lanes>=16: K=k0+8..15,24..31 at N=lane-16.
__device__ __forceinline__ v16bf load_bt_frag(const bf16_t* src, int n0, int ldm,
                                              int k0, int lane) {
  int n  = n0 + (lane & 15);
  int ko = k0 + ((lane & 16) ? 8 : 0);
  const bf16_t* p = src + (size_t)n * ldm + ko;
  FragU f;
  f.u[0] = *(const u32x4*)(p);
  f.u[1] = *(const u32x4*)(p + 16);
  return f.v;
}

// ---- CDNA5 async global->LDS copy (ASYNCcnt path) ----
__device__ __forceinline__ unsigned lds_off_u32(const void* p) {
  // generic pointers into LDS carry the ds offset in the low 32 bits
  return (unsigned)(unsigned long long)p;
}
__device__ __forceinline__ void async_ld16(unsigned lds_off, const bf16_t* g) {
  asm volatile("global_load_async_to_lds_b128 %0, %1, off"
               :: "v"(lds_off), "v"(g) : "memory");
}
__device__ __forceinline__ void wait_async0() {
  asm volatile("s_wait_asynccnt 0x0" ::: "memory");
}

// ---------------- fp32 -> bf16 conversion (vectorized x4) ----------------
__global__ void cvt_f32_to_bf16_k(const float* __restrict__ in,
                                  bf16_t* __restrict__ out, int n4) {
  int i = blockIdx.x * blockDim.x + threadIdx.x;
  if (i >= n4) return;
  f32x4 f = ((const f32x4*)in)[i];
  union { bf16_t h[4]; u32x2 u; } o;
  o.h[0] = (bf16_t)f[0]; o.h[1] = (bf16_t)f[1];
  o.h[2] = (bf16_t)f[2]; o.h[3] = (bf16_t)f[3];
  ((u32x2*)out)[i] = o.u;
}

// ---------------- QKV GEMM + RMSnorm + RoPE epilogue ----------------
// Wave tile: 64 rows x 64 cols (one full head of q/k/v). Block = 4 waves in M,
// sharing one async-staged 64x32 weight tile per K-step (double buffered).
// 16 WMMAs per K-step for 8 global + 8 LDS fragment loads.
__global__ void __launch_bounds__(128)
qkv_rms_rope_k(const bf16_t* __restrict__ xb, const bf16_t* __restrict__ wqkv,
               const float* __restrict__ fc, const float* __restrict__ fs,
               const float* __restrict__ qg, const float* __restrict__ kg,
               bf16_t* __restrict__ Qb, bf16_t* __restrict__ Kb,
               bf16_t* __restrict__ Vt) {
  __shared__ __align__(16) bf16_t Bs[2][64 * 32];
  const int tid  = threadIdx.x;
  const int lane = tid & 31;
  const int wave = tid >> 5;
  const int hi   = (lane >> 4) & 1;
  const int lc   = lane & 15;
  const int m0   = blockIdx.x * 256 + wave * 64;
  const int d0   = blockIdx.y * 64;

  auto stageB = [&](int buf, int k0) {
#pragma unroll
    for (int c = 0; c < 2; ++c) {
      int q = c * 128 + tid;            // 0..255 chunks of 16B
      int row = q >> 2, part = q & 3;   // 64 rows x 4 chunks
      async_ld16(lds_off_u32(&Bs[buf][row * 32 + part * 8]),
                 wqkv + (size_t)(d0 + row) * kC + k0 + part * 8);
    }
  };

  v8f zero = {};
  v8f acc[4][4];
#pragma unroll
  for (int mt = 0; mt < 4; ++mt)
#pragma unroll
    for (int j = 0; j < 4; ++j) acc[mt][j] = zero;

  stageB(0, 0);
  for (int it = 0; it < kC / 32; ++it) {
    const int k0 = it * 32, buf = it & 1;
    wait_async0();
    __syncthreads();
    if (it + 1 < kC / 32) stageB(buf ^ 1, k0 + 32);
    v16bf a[4];
#pragma unroll
    for (int mt = 0; mt < 4; ++mt)
      a[mt] = load_a_frag(xb, m0 + mt * 16, kC, k0, lane);
#pragma unroll
    for (int j = 0; j < 4; ++j) {
      v16bf b = load_bt_frag(Bs[buf], j * 16, 32, 0, lane);
#pragma unroll
      for (int mt = 0; mt < 4; ++mt)
        acc[mt][j] = wmma_bf16(a[mt], b, acc[mt][j]);
    }
  }

  if (d0 >= 2 * kC) {  // ---- V: store transposed [b][h][hd][n] ----
    int h = (d0 - 2 * kC) >> 6;
#pragma unroll
    for (int mt = 0; mt < 4; ++mt)
#pragma unroll
      for (int r = 0; r < 8; ++r) {
        int m = m0 + mt * 16 + r + hi * 8;
        int b = m >> 11, n = m & (kN - 1);
#pragma unroll
        for (int j = 0; j < 4; ++j) {
          int dm = j * 16 + lc;
          Vt[(((size_t)(b * kH + h)) * kHd + dm) * kN + n] = (bf16_t)acc[mt][j][r];
        }
      }
    return;
  }

  // ---- Q/K: RMS norm over hd=64, RoPE, store [b][h][n][hd] ----
  const bool  isQ   = d0 < kC;
  const float* gsrc = isQ ? qg : kg;
  const float scale = isQ ? 0.125f : 1.0f;  // fold hd^-0.5 into Q
  const int   h     = (isQ ? d0 : d0 - kC) >> 6;
  bf16_t* dst       = isQ ? Qb : Kb;
  float g[4];
#pragma unroll
  for (int j = 0; j < 4; ++j) g[j] = gsrc[j * 16 + lc];

#pragma unroll
  for (int mt = 0; mt < 4; ++mt) {
#pragma unroll
    for (int r = 0; r < 8; ++r) {
      int m = m0 + mt * 16 + r + hi * 8;
      float vals[4], ss = 0.f;
#pragma unroll
      for (int j = 0; j < 4; ++j) { vals[j] = acc[mt][j][r]; ss += vals[j] * vals[j]; }
      ss += __shfl_xor(ss, 1, 32);
      ss += __shfl_xor(ss, 2, 32);
      ss += __shfl_xor(ss, 4, 32);
      ss += __shfl_xor(ss, 8, 32);          // full row sum within 16-lane group
      float inv = 8.0f / fmaxf(sqrtf(ss), 1e-12f);
      int b = m >> 11, n = m & (kN - 1);
      const float* fcr = fc + (size_t)m * kHd;
      const float* fsr = fs + (size_t)m * kHd;
      bf16_t* drow = dst + (((size_t)(b * kH + h)) * kN + n) * kHd;
#pragma unroll
      for (int j = 0; j < 4; ++j) {
        int dm = j * 16 + lc;
        float v    = vals[j] * inv * g[j];
        float part = __shfl_xor(v, 1, 32);  // RoPE pair partner (dm ^ 1)
        float c = fcr[dm], s = fsr[dm];
        float o = (dm & 1) ? (v * c + part * s) : (v * c - part * s);
        drow[dm] = (bf16_t)(o * scale);
      }
    }
  }
}

// ---------------- Flash attention (S^T formulation) ----------------
// Wave: 32 queries (2 q-tiles) x full hd. S^T = K·Q^T so softmax stats are
// per-lane and the S^T accumulator registers ARE the A-fragment layout of P for
// the P·V WMMA. K/V tiles identical for all waves -> async-staged in LDS and
// their fragments shared by both q-tiles (1 ds-load per WMMA).
__global__ void __launch_bounds__(128)
flash_attn_k(const bf16_t* __restrict__ Qb, const bf16_t* __restrict__ Kb,
             const bf16_t* __restrict__ Vt, const float* __restrict__ mask,
             bf16_t* __restrict__ Ob) {
  __shared__ __align__(16) bf16_t Ks[2][64 * 64];  // [key_local][d]
  __shared__ __align__(16) bf16_t Vs[2][64 * 64];  // [d][key_local]
  const int tid  = threadIdx.x;
  const int lane = tid & 31;
  const int wave = tid >> 5;
  const int hi   = (lane >> 4) & 1;
  const int lc   = lane & 15;
  const int bh   = blockIdx.y;
  const int b    = bh >> 4;
  const int h    = bh & 15;
  const int q0w  = blockIdx.x * 128 + wave * 32;   // 32 queries per wave

  const bf16_t* Qh = Qb + (size_t)bh * kN * kHd;
  const bf16_t* Kh = Kb + (size_t)bh * kN * kHd;
  const bf16_t* Vh = Vt + (size_t)bh * kHd * kN;

  auto stageKV = [&](int buf, int kb) {
#pragma unroll
    for (int c = 0; c < 4; ++c) {
      int q = c * 128 + tid;            // 0..511 chunks of 16B per matrix
      int row = q >> 3, part = q & 7;   // 64 rows x 8 chunks (128B rows)
      async_ld16(lds_off_u32(&Ks[buf][row * 64 + part * 8]),
                 Kh + (size_t)(kb + row) * kHd + part * 8);
      async_ld16(lds_off_u32(&Vs[buf][row * 64 + part * 8]),
                 Vh + (size_t)row * kN + kb + part * 8);
    }
  };

  v8f zero = {};
  v16bf bq[2][2];                       // [q-tile][k-half] Q^T frags, loaded once
#pragma unroll
  for (int qt = 0; qt < 2; ++qt) {
    bq[qt][0] = load_bt_frag(Qh, q0w + qt * 16, kHd, 0,  lane);
    bq[qt][1] = load_bt_frag(Qh, q0w + qt * 16, kHd, 32, lane);
  }

  v8f ot[2][4];
#pragma unroll
  for (int qt = 0; qt < 2; ++qt)
#pragma unroll
    for (int j = 0; j < 4; ++j) ot[qt][j] = zero;
  float mrun[2] = {-3.0e38f, -3.0e38f}, lrun[2] = {0.f, 0.f};
  const float* mrow[2];
#pragma unroll
  for (int qt = 0; qt < 2; ++qt)
    mrow[qt] = mask + (size_t)(q0w + qt * 16 + lc) * kN;

  stageKV(0, 0);
  for (int it = 0; it < kN / 64; ++it) {
    const int kb = it * 64, buf = it & 1;
    wait_async0();
    __syncthreads();
    if (it + 1 < kN / 64) stageKV(buf ^ 1, kb + 64);
    const bf16_t* Kt  = Ks[buf];
    const bf16_t* Vb2 = Vs[buf];

    v8f st[2][4];
#pragma unroll
    for (int kt = 0; kt < 4; ++kt) {
      v16bf a0 = load_a_frag(Kt, kt * 16, kHd, 0,  lane);
      v16bf a1 = load_a_frag(Kt, kt * 16, kHd, 32, lane);
#pragma unroll
      for (int qt = 0; qt < 2; ++qt) {
        st[qt][kt] = wmma_bf16(a1, bq[qt][1], wmma_bf16(a0, bq[qt][0], zero));
        const f32x4* mp = (const f32x4*)(mrow[qt] + kb + kt * 16 + hi * 8);
        f32x4 ma = mp[0], mb2 = mp[1];
        st[qt][kt][0] += ma[0];  st[qt][kt][1] += ma[1];
        st[qt][kt][2] += ma[2];  st[qt][kt][3] += ma[3];
        st[qt][kt][4] += mb2[0]; st[qt][kt][5] += mb2[1];
        st[qt][kt][6] += mb2[2]; st[qt][kt][7] += mb2[3];
      }
    }

    v16bf pa[2][2];
    float alpha[2];
#pragma unroll
    for (int qt = 0; qt < 2; ++qt) {
      // online softmax over this 64-key block (q = lc per lane)
      float bm = -3.0e38f;
#pragma unroll
      for (int kt = 0; kt < 4; ++kt)
#pragma unroll
        for (int r = 0; r < 8; ++r) bm = fmaxf(bm, st[qt][kt][r]);
      bm = fmaxf(bm, __shfl_xor(bm, 16, 32));
      float mnew = fmaxf(mrun[qt], bm);
      alpha[qt]  = __expf(mrun[qt] - mnew);
      float psum = 0.f;
#pragma unroll
      for (int kt = 0; kt < 4; ++kt)
#pragma unroll
        for (int r = 0; r < 8; ++r) {
          float p = __expf(st[qt][kt][r] - mnew);
          st[qt][kt][r] = p;
          psum += p;
        }
      psum += __shfl_xor(psum, 16, 32);
      lrun[qt] = lrun[qt] * alpha[qt] + psum;
      mrun[qt] = mnew;
      // P (bf16) A-fragments directly from S^T accumulators — no transpose
#pragma unroll
      for (int e = 0; e < 8; ++e) {
        pa[qt][0][e]     = (bf16_t)st[qt][0][e];
        pa[qt][0][8 + e] = (bf16_t)st[qt][1][e];
        pa[qt][1][e]     = (bf16_t)st[qt][2][e];
        pa[qt][1][8 + e] = (bf16_t)st[qt][3][e];
      }
    }
    float al[2][8];
#pragma unroll
    for (int qt = 0; qt < 2; ++qt)
#pragma unroll
      for (int r = 0; r < 8; ++r) al[qt][r] = __shfl(alpha[qt], r + hi * 8, 32);
#pragma unroll
    for (int j = 0; j < 4; ++j) {
      v16bf bv0 = load_bt_frag(Vb2, j * 16, kHd, 0,  lane);
      v16bf bv1 = load_bt_frag(Vb2, j * 16, kHd, 32, lane);
#pragma unroll
      for (int qt = 0; qt < 2; ++qt) {
#pragma unroll
        for (int r = 0; r < 8; ++r) ot[qt][j][r] *= al[qt][r];
        ot[qt][j] = wmma_bf16(pa[qt][1], bv1, wmma_bf16(pa[qt][0], bv0, ot[qt][j]));
      }
    }
  }

#pragma unroll
  for (int qt = 0; qt < 2; ++qt) {
    float li[8];
#pragma unroll
    for (int r = 0; r < 8; ++r) li[r] = __shfl(lrun[qt], r + hi * 8, 32);
#pragma unroll
    for (int j = 0; j < 4; ++j)
#pragma unroll
      for (int r = 0; r < 8; ++r) {
        int q = q0w + qt * 16 + r + hi * 8;
        float v = ot[qt][j][r] / li[r];
        Ob[((size_t)(b * kN + q)) * kC + h * kHd + j * 16 + lc] = (bf16_t)v;
      }
  }
}

// ---------------- Projection GEMM + bias ----------------
__global__ void __launch_bounds__(128)
proj_bias_k(const bf16_t* __restrict__ Ob, const bf16_t* __restrict__ wp,
            const float* __restrict__ bias, float* __restrict__ out) {
  __shared__ __align__(16) bf16_t Bs[2][64 * 32];
  const int tid  = threadIdx.x;
  const int lane = tid & 31;
  const int wave = tid >> 5;
  const int hi   = (lane >> 4) & 1;
  const int lc   = lane & 15;
  const int m0   = blockIdx.x * 256 + wave * 64;
  const int d0   = blockIdx.y * 64;

  auto stageB = [&](int buf, int k0) {
#pragma unroll
    for (int c = 0; c < 2; ++c) {
      int q = c * 128 + tid;
      int row = q >> 2, part = q & 3;
      async_ld16(lds_off_u32(&Bs[buf][row * 32 + part * 8]),
                 wp + (size_t)(d0 + row) * kC + k0 + part * 8);
    }
  };

  v8f zero = {};
  v8f acc[4][4];
#pragma unroll
  for (int mt = 0; mt < 4; ++mt)
#pragma unroll
    for (int j = 0; j < 4; ++j) acc[mt][j] = zero;

  stageB(0, 0);
  for (int it = 0; it < kC / 32; ++it) {
    const int k0 = it * 32, buf = it & 1;
    wait_async0();
    __syncthreads();
    if (it + 1 < kC / 32) stageB(buf ^ 1, k0 + 32);
    v16bf a[4];
#pragma unroll
    for (int mt = 0; mt < 4; ++mt)
      a[mt] = load_a_frag(Ob, m0 + mt * 16, kC, k0, lane);
#pragma unroll
    for (int j = 0; j < 4; ++j) {
      v16bf b = load_bt_frag(Bs[buf], j * 16, 32, 0, lane);
#pragma unroll
      for (int mt = 0; mt < 4; ++mt)
        acc[mt][j] = wmma_bf16(a[mt], b, acc[mt][j]);
    }
  }

  float bs[4];
#pragma unroll
  for (int j = 0; j < 4; ++j) bs[j] = bias[d0 + j * 16 + lc];
#pragma unroll
  for (int mt = 0; mt < 4; ++mt)
#pragma unroll
    for (int r = 0; r < 8; ++r) {
      int m = m0 + mt * 16 + r + hi * 8;
#pragma unroll
      for (int j = 0; j < 4; ++j)
        out[(size_t)m * kC + d0 + j * 16 + lc] = acc[mt][j][r] + bs[j];
    }
}

extern "C" void kernel_launch(void* const* d_in, const int* in_sizes, int n_in,
                              void* d_out, int out_size, void* d_ws, size_t ws_size,
                              hipStream_t stream) {
  const float* x      = (const float*)d_in[0];
  const float* fc     = (const float*)d_in[1];
  const float* fs     = (const float*)d_in[2];
  const float* mask   = (const float*)d_in[3];
  const float* w_qkv  = (const float*)d_in[4];
  const float* w_proj = (const float*)d_in[5];
  const float* b_proj = (const float*)d_in[6];
  const float* q_g    = (const float*)d_in[7];
  const float* k_g    = (const float*)d_in[8];
  float* out = (float*)d_out;

  char* ws = (char*)d_ws;
  size_t off = 0;
  bf16_t* xb  = (bf16_t*)(ws + off); off += (size_t)kM * kC * 2;        // 8 MB
  bf16_t* wqb = (bf16_t*)(ws + off); off += (size_t)k3C * kC * 2;       // 6 MB
  bf16_t* wpb = (bf16_t*)(ws + off); off += (size_t)kC * kC * 2;        // 2 MB
  bf16_t* Qb  = (bf16_t*)(ws + off); off += (size_t)kM * kC * 2;        // 8 MB
  bf16_t* Kb  = (bf16_t*)(ws + off); off += (size_t)kM * kC * 2;        // 8 MB
  bf16_t* Vt  = (bf16_t*)(ws + off); off += (size_t)kM * kC * 2;        // 8 MB
  bf16_t* Ob  = (bf16_t*)(ws + off); off += (size_t)kM * kC * 2;        // 8 MB

  cvt_f32_to_bf16_k<<<dim3((kM * kC / 4 + 255) / 256), dim3(256), 0, stream>>>(
      x, xb, kM * kC / 4);
  cvt_f32_to_bf16_k<<<dim3((k3C * kC / 4 + 255) / 256), dim3(256), 0, stream>>>(
      w_qkv, wqb, k3C * kC / 4);
  cvt_f32_to_bf16_k<<<dim3((kC * kC / 4 + 255) / 256), dim3(256), 0, stream>>>(
      w_proj, wpb, kC * kC / 4);

  qkv_rms_rope_k<<<dim3(kM / 256, k3C / 64), dim3(128), 0, stream>>>(
      xb, wqb, fc, fs, q_g, k_g, Qb, Kb, Vt);

  flash_attn_k<<<dim3(kN / 128, 32), dim3(128), 0, stream>>>(Qb, Kb, Vt, mask, Ob);

  proj_bias_k<<<dim3(kM / 256, kC / 64), dim3(128), 0, stream>>>(Ob, wpb, b_proj, out);
}
